// Attention_42803644072865
// MI455X (gfx1250) — matrix operand order repaired and checked by reading, not compiled, not run on hardware
//
#include <hip/hip_runtime.h>
#include <hip/hip_fp16.h>

typedef _Float16 v16h __attribute__((ext_vector_type(16)));
typedef _Float16 v8h  __attribute__((ext_vector_type(8)));
typedef _Float16 v4h  __attribute__((ext_vector_type(4)));
typedef float    v8f  __attribute__((ext_vector_type(8)));

#define WMMA_F16(a, b, c) \
    __builtin_amdgcn_wmma_f32_16x16x32_f16(false, (a), false, (b), (short)0, (c), false, false)

constexpr int Bc = 2, Sc = 2048, Dc = 1024, Hc = 16, HDc = 64;
constexpr int Nqkv = Hc * HDc; // 1024

static __device__ __forceinline__ v16h make_v16h(const v8h lo, const v8h hi) {
    v16h r;
    #pragma unroll
    for (int i = 0; i < 8; ++i) { r[i] = lo[i]; r[i + 8] = hi[i]; }
    return r;
}

// ---------------------------------------------------------------------------
// Prep A: elementwise fp32 -> f16 (for X)
// ---------------------------------------------------------------------------
__global__ __launch_bounds__(256) void cvt_f16_kernel(
    const float* __restrict__ src, _Float16* __restrict__ dst)
{
    const size_t i4 = ((size_t)blockIdx.x * 256 + threadIdx.x) * 4;
    const float4 v = *(const float4*)(src + i4);
    v4h h;
    h[0] = (_Float16)v.x; h[1] = (_Float16)v.y;
    h[2] = (_Float16)v.z; h[3] = (_Float16)v.w;
    *(v4h*)(dst + i4) = h;
}

// ---------------------------------------------------------------------------
// Prep B: transpose + convert a 1024x1024 fp32 weight (K-major) into f16
// N-major (dst[n][k]).  The 32x32 fp32 tile is filled with the gfx1250
// async global->LDS mover (ASYNCcnt-tracked), then read back transposed.
// Row stride 36 floats (144 B) keeps every async b128 store 16B-aligned.
// ---------------------------------------------------------------------------
__global__ __launch_bounds__(256) void transpose_w_kernel(
    const float* __restrict__ src, _Float16* __restrict__ dst)
{
    __shared__ float tile[32][36];    // sole LDS allocation -> starts at offset 0
    const int tid = threadIdx.x;
    const int r  = tid >> 3;          // 0..31
    const int c4 = (tid & 7) * 4;     // 0,4,...,28
    const int kb = blockIdx.x * 32;   // K tile base (src row)
    const int nb = blockIdx.y * 32;   // N tile base (src col)

    // async copy: 16 bytes per lane, global (coalesced rows) -> LDS tile
    {
        const unsigned long long gaddr =
            (unsigned long long)(src + (size_t)(kb + r) * Dc + nb + c4);
        const unsigned ldsOff = (unsigned)((r * 36 + c4) * sizeof(float));
        asm volatile("global_load_async_to_lds_b128 %0, %1, off"
                     :: "v"(ldsOff), "v"(gaddr) : "memory");
        asm volatile("s_wait_asynccnt 0x0" ::: "memory");
    }
    __syncthreads();

    v4h h;
    #pragma unroll
    for (int j = 0; j < 4; ++j) h[j] = (_Float16)tile[c4 + j][r];
    *(v4h*)(dst + (size_t)(nb + r) * Dc + kb + c4) = h;
}

// ---------------------------------------------------------------------------
// Kernel 1: fused QKV projection.  Wave tile = 16 (B*S rows) x 64 cols of the
// concatenated [Q|K|V] output (N=3072).  A from pre-converted f16 X rows,
// B from transposed f16 weights -- every fragment is 2x global_load_b128.
// Q,K stored [B,H,S,HD]; V stored head-transposed [B,H,HD,S].
// ---------------------------------------------------------------------------
__global__ __launch_bounds__(128) void qkv_proj_kernel(
    const _Float16* __restrict__ Xh, const _Float16* __restrict__ WqkvT,
    const float* __restrict__ bq, const float* __restrict__ bk,
    const float* __restrict__ bv,
    _Float16* __restrict__ Qh, _Float16* __restrict__ Kh, _Float16* __restrict__ Vt)
{
    const int tid  = threadIdx.x;
    const int wave = tid >> 5;
    const int lane = tid & 31;
    const int l16  = lane & 15;
    const int half = lane >> 4;

    const int mbase = blockIdx.x * 16;               // rows over B*S
    const int ngl   = blockIdx.y * 256 + wave * 64;  // 0..3071 over [Q|K|V]
    const int which = ngl >> 10;                     // 0=Q 1=K 2=V
    const float* bias = (which == 0) ? bq : ((which == 1) ? bk : bv);

    const _Float16* arow = Xh + (size_t)(mbase + l16) * Dc;
    const _Float16* brow[4];
    #pragma unroll
    for (int j = 0; j < 4; ++j)
        brow[j] = WqkvT + (size_t)(ngl + j * 16 + l16) * Dc;

    v8f acc[4] = {v8f{}, v8f{}, v8f{}, v8f{}};
    for (int k0 = 0; k0 < Dc; k0 += 32) {
        const v16h afrag = make_v16h(*(const v8h*)(arow + k0 + half * 16),
                                     *(const v8h*)(arow + k0 + half * 16 + 8));
        if (k0 + 32 < Dc) {
            __builtin_prefetch(arow + k0 + 32 + half * 8, 0, 3);
            __builtin_prefetch(brow[0] + k0 + 32 + half * 16, 0, 3);
        }
        #pragma unroll
        for (int j = 0; j < 4; ++j) {
            const v16h bfrag = make_v16h(*(const v8h*)(brow[j] + k0 + half * 16),
                                         *(const v8h*)(brow[j] + k0 + half * 16 + 8));
            acc[j] = WMMA_F16(afrag, bfrag, acc[j]);
        }
    }

    #pragma unroll
    for (int j = 0; j < 4; ++j) {
        const int ncol = (ngl & 1023) + j * 16 + l16;
        const float bval = bias[ncol];
        const int hh = ncol >> 6, hd = ncol & 63;
        #pragma unroll
        for (int r = 0; r < 8; ++r) {
            const int row = mbase + r + half * 8;    // C/D layout: m = r + 8*half
            const int b = row / Sc, s = row % Sc;
            const _Float16 val = (_Float16)(acc[j][r] + bval);
            if (which == 0)
                Qh[(((size_t)b * Hc + hh) * Sc + s) * HDc + hd] = val;
            else if (which == 1)
                Kh[(((size_t)b * Hc + hh) * Sc + s) * HDc + hd] = val;
            else
                Vt[(((size_t)b * Hc + hh) * HDc + hd) * Sc + s] = val;
        }
    }
}

// ---------------------------------------------------------------------------
// Kernel 2: flash attention.  One wave per (b, h, 16-query tile); online
// softmax over 64 key tiles of 32 keys.  All K/V fragments are contiguous
// b128 loads; P re-swizzled C/D -> A layout through *wave-private* LDS.
// Same-wave DS ops are in-order on CDNA5, so only a compiler fence
// (wave_barrier) is needed -- no hardware workgroup barriers in the loop.
// ---------------------------------------------------------------------------
__global__ __launch_bounds__(128) void attn_kernel(
    const _Float16* __restrict__ Qh, const _Float16* __restrict__ Kh,
    const _Float16* __restrict__ Vt, const float* __restrict__ mask,
    _Float16* __restrict__ Ctx)
{
    __shared__ _Float16 ldsP[4][16 * 32];

    const int tid  = threadIdx.x;
    const int wave = tid >> 5;
    const int lane = tid & 31;
    const int l16  = lane & 15;
    const int half = lane >> 4;

    const int flat  = blockIdx.x * 4 + wave;   // B*H*(S/16) = 4096
    const int qt    = flat & 127;
    const int hh    = (flat >> 7) & 15;
    const int b     = flat >> 11;
    const int qbase = qt * 16;

    const _Float16* Qb = Qh + ((size_t)b * Hc + hh) * Sc * HDc;
    const _Float16* Kb = Kh + ((size_t)b * Hc + hh) * Sc * HDc;
    const _Float16* Vb = Vt + ((size_t)b * Hc + hh) * (size_t)HDc * Sc; // [HD][S]
    const float*    mb = mask + (size_t)b * Sc;

    // Q fragments (K-dim 64 -> two 16x32 A tiles), pre-scaled by 1/sqrt(HD)
    v16h aq0, aq1;
    {
        const _Float16* qrow = Qb + (size_t)(qbase + l16) * HDc;
        const v8h q00 = *(const v8h*)(qrow + half * 16);
        const v8h q01 = *(const v8h*)(qrow + half * 16 + 8);
        const v8h q10 = *(const v8h*)(qrow + 32 + half * 16);
        const v8h q11 = *(const v8h*)(qrow + 32 + half * 16 + 8);
        const _Float16 qs = (_Float16)0.125f;  // 1/sqrt(64), exact
        #pragma unroll
        for (int i = 0; i < 8; ++i) {
            aq0[i] = q00[i] * qs; aq0[i + 8] = q01[i] * qs;
            aq1[i] = q10[i] * qs; aq1[i + 8] = q11[i] * qs;
        }
    }

    float rowmax[8], rowsum[8];
    v8f ctxa[4] = {v8f{}, v8f{}, v8f{}, v8f{}};
    #pragma unroll
    for (int r = 0; r < 8; ++r) { rowmax[r] = -3.0e38f; rowsum[r] = 0.0f; }

    for (int kt = 0; kt < Sc / 32; ++kt) {
        const int keyb = kt * 32;

        // ---- scores: 16 queries x 32 keys = two 16x16 WMMA tiles ----
        v8f scr[2];
        #pragma unroll
        for (int sub = 0; sub < 2; ++sub) {
            const _Float16* krow = Kb + (size_t)(keyb + sub * 16 + l16) * HDc;
            const v16h bk0 = make_v16h(*(const v8h*)(krow + half * 16),
                                       *(const v8h*)(krow + half * 16 + 8));
            const v16h bk1 = make_v16h(*(const v8h*)(krow + 32 + half * 16),
                                       *(const v8h*)(krow + 32 + half * 16 + 8));
            v8f d = {};
            d = WMMA_F16(aq0, bk0, d);
            d = WMMA_F16(aq1, bk1, d);
            const float mv  = mb[keyb + sub * 16 + l16];
            const float pen = 1.0e6f * (1.0f - mv);
            #pragma unroll
            for (int r = 0; r < 8; ++r) d[r] -= pen;
            scr[sub] = d;
        }

        // ---- online softmax (each row lives in one 16-lane half) ----
        float cand[8];
        #pragma unroll
        for (int r = 0; r < 8; ++r) cand[r] = fmaxf(scr[0][r], scr[1][r]);
        #pragma unroll
        for (int off = 1; off < 16; off <<= 1) {
            #pragma unroll
            for (int r = 0; r < 8; ++r)
                cand[r] = fmaxf(cand[r], __shfl_xor(cand[r], off, 32));
        }

        float fac[8], psum[8];
        #pragma unroll
        for (int r = 0; r < 8; ++r) {
            const float nm = fmaxf(rowmax[r], cand[r]);
            fac[r]    = __expf(rowmax[r] - nm);
            rowmax[r] = nm;
            const float p0 = __expf(scr[0][r] - nm);
            const float p1 = __expf(scr[1][r] - nm);
            ldsP[wave][(r + half * 8) * 32 + l16]      = (_Float16)p0;
            ldsP[wave][(r + half * 8) * 32 + 16 + l16] = (_Float16)p1;
            psum[r] = p0 + p1;
        }
        #pragma unroll
        for (int off = 1; off < 16; off <<= 1) {
            #pragma unroll
            for (int r = 0; r < 8; ++r) psum[r] += __shfl_xor(psum[r], off, 32);
        }
        #pragma unroll
        for (int r = 0; r < 8; ++r) {
            rowsum[r] = rowsum[r] * fac[r] + psum[r];
            ctxa[0][r] *= fac[r]; ctxa[1][r] *= fac[r];
            ctxa[2][r] *= fac[r]; ctxa[3][r] *= fac[r];
        }

        // wave-private LDS swizzle: DS in-order within a wave, only need to
        // stop the compiler from moving the reads above the writes.
        __builtin_amdgcn_wave_barrier();

        // ---- ctx[16x64] += P[16x32] * V[32x64] ----
        const v16h pa = make_v16h(*(const v8h*)&ldsP[wave][l16 * 32 + half * 16],
                                  *(const v8h*)&ldsP[wave][l16 * 32 + half * 16 + 8]);
        #pragma unroll
        for (int t = 0; t < 4; ++t) {
            // B layout: lane col n = t*16+l16 (= hd), slots = keys (contiguous in Vt)
            const _Float16* vrow = Vb + (size_t)(t * 16 + l16) * Sc + keyb + half * 16;
            const v16h bvf = make_v16h(*(const v8h*)(vrow), *(const v8h*)(vrow + 8));
            ctxa[t] = WMMA_F16(pa, bvf, ctxa[t]);
        }
        __builtin_amdgcn_wave_barrier();   // WAR: next iter rewrites ldsP
    }

    // normalize and store ctx f16 into [B, S, H*HD]
    #pragma unroll
    for (int r = 0; r < 8; ++r) {
        const float inv = 1.0f / rowsum[r];
        const int   row = qbase + r + half * 8;
        _Float16* cp = Ctx + ((size_t)b * Sc + row) * Nqkv + hh * HDc + l16;
        cp[0]  = (_Float16)(ctxa[0][r] * inv);
        cp[16] = (_Float16)(ctxa[1][r] * inv);
        cp[32] = (_Float16)(ctxa[2][r] * inv);
        cp[48] = (_Float16)(ctxa[3][r] * inv);
    }
}

// ---------------------------------------------------------------------------
// Kernel 3: output projection  Out[B*S,1024] = Ctx(f16) @ Wo + bo  (fp32 out)
// Wave tile 16x64, B fragments from transposed f16 WoT.
// ---------------------------------------------------------------------------
__global__ __launch_bounds__(128) void out_proj_kernel(
    const _Float16* __restrict__ Ctx, const _Float16* __restrict__ WoT,
    const float* __restrict__ bo, float* __restrict__ Out)
{
    const int tid  = threadIdx.x;
    const int wave = tid >> 5;
    const int lane = tid & 31;
    const int l16  = lane & 15;
    const int half = lane >> 4;

    const int mbase = blockIdx.x * 16;
    const int nbase = blockIdx.y * 256 + wave * 64;

    const _Float16* arow = Ctx + (size_t)(mbase + l16) * Nqkv;
    const _Float16* brow[4];
    #pragma unroll
    for (int j = 0; j < 4; ++j)
        brow[j] = WoT + (size_t)(nbase + j * 16 + l16) * Nqkv;

    v8f acc[4] = {v8f{}, v8f{}, v8f{}, v8f{}};
    for (int k0 = 0; k0 < Nqkv; k0 += 32) {
        const v16h afrag = make_v16h(*(const v8h*)(arow + k0 + half * 16),
                                     *(const v8h*)(arow + k0 + half * 16 + 8));
        if (k0 + 32 < Nqkv) {
            __builtin_prefetch(arow + k0 + 32 + half * 8, 0, 3);
            __builtin_prefetch(brow[0] + k0 + 32 + half * 16, 0, 3);
        }
        #pragma unroll
        for (int j = 0; j < 4; ++j) {
            const v16h bfrag = make_v16h(*(const v8h*)(brow[j] + k0 + half * 16),
                                         *(const v8h*)(brow[j] + k0 + half * 16 + 8));
            acc[j] = WMMA_F16(afrag, bfrag, acc[j]);
        }
    }

    #pragma unroll
    for (int j = 0; j < 4; ++j) {
        const float bias = bo[nbase + j * 16 + l16];
        #pragma unroll
        for (int r = 0; r < 8; ++r)
            Out[(size_t)(mbase + r + half * 8) * Dc + nbase + j * 16 + l16] =
                acc[j][r] + bias;
    }
}

// ---------------------------------------------------------------------------
extern "C" void kernel_launch(void* const* d_in, const int* in_sizes, int n_in,
                              void* d_out, int out_size, void* d_ws, size_t ws_size,
                              hipStream_t stream) {
    (void)in_sizes; (void)n_in; (void)out_size; (void)ws_size;
    const float* X    = (const float*)d_in[0];
    const float* mask = (const float*)d_in[1];
    const float* Wq   = (const float*)d_in[2];
    const float* bq   = (const float*)d_in[3];
    const float* Wk   = (const float*)d_in[4];
    const float* bk   = (const float*)d_in[5];
    const float* Wv   = (const float*)d_in[6];
    const float* bv   = (const float*)d_in[7];
    const float* Wo   = (const float*)d_in[8];
    const float* bo   = (const float*)d_in[9];
    float* Out = (float*)d_out;

    const size_t MW = (size_t)Dc * Nqkv;            // 1M elements per matrix
    const size_t per = (size_t)Bc * Hc * Sc * HDc;  // 4M f16 elements
    _Float16* Xh    = (_Float16*)d_ws;              //  8 MB
    _Float16* WqkvT = Xh + (size_t)Bc * Sc * Dc;    //  6 MB (Q|K|V transposed)
    _Float16* WoT   = WqkvT + 3 * MW;               //  2 MB
    _Float16* Qh    = WoT + MW;                     //  8 MB
    _Float16* Kh    = Qh + per;                     //  8 MB
    _Float16* Vt    = Kh + per;                     //  8 MB  [B,H,HD,S]
    _Float16* Ct    = Vt + per;                     //  8 MB  -> 48 MB total

    const dim3 b256(256), b128(128);
    // prep: f16 conversion + weight transposes
    cvt_f16_kernel<<<dim3((Bc * Sc * Dc) / 1024), b256, 0, stream>>>(X, Xh);
    transpose_w_kernel<<<dim3(32, 32), b256, 0, stream>>>(Wq, WqkvT + 0 * MW);
    transpose_w_kernel<<<dim3(32, 32), b256, 0, stream>>>(Wk, WqkvT + 1 * MW);
    transpose_w_kernel<<<dim3(32, 32), b256, 0, stream>>>(Wv, WqkvT + 2 * MW);
    transpose_w_kernel<<<dim3(32, 32), b256, 0, stream>>>(Wo, WoT);
    // compute
    qkv_proj_kernel<<<dim3((Bc * Sc) / 16, (3 * Nqkv) / 256), b128, 0, stream>>>(
        Xh, WqkvT, bq, bk, bv, Qh, Kh, Vt);
    attn_kernel<<<dim3((Bc * Hc * (Sc / 16)) / 4), b128, 0, stream>>>(
        Qh, Kh, Vt, mask, Ct);
    out_proj_kernel<<<dim3((Bc * Sc) / 16, Dc / 256), b128, 0, stream>>>(
        Ct, WoT, bo, Out);
}
